// RelativeMultiHeadAttention_46273977647184
// MI455X (gfx1250) — matrix operand order repaired and checked by reading, not compiled, not run on hardware
//
#include <hip/hip_runtime.h>
#include <hip/hip_bf16.h>
#include <math.h>

typedef __attribute__((ext_vector_type(16))) _Float16 v16h;
typedef __attribute__((ext_vector_type(8)))  _Float16 v8h;
typedef __attribute__((ext_vector_type(4)))  _Float16 v4h;
typedef __attribute__((ext_vector_type(8)))  float    v8f;

#define D_MODEL   1024
#define NUM_HEADS 16
#define HD        64
#define SEQ       2048
#define BATCH     4
#define MTOT      (BATCH * SEQ)   // 8192 rows

// ---------------------------------------------------------------------------
// CDNA5 async copy: global memory -> LDS, 16B per lane, tracked by ASYNCcnt.
// ---------------------------------------------------------------------------
__device__ __forceinline__ void async_ld_b128(void* lds, const void* gaddr) {
  unsigned l = (unsigned)(size_t)lds;
  asm volatile("global_load_async_to_lds_b128 %0, %1, off"
               :: "v"(l), "v"(gaddr) : "memory");
}
__device__ __forceinline__ void wait_async0() {
  asm volatile("s_wait_asynccnt 0x0" ::: "memory");
}
__device__ __forceinline__ void async_row16(_Float16* ldsDst,
                                            const _Float16* gSrc) {
  async_ld_b128(ldsDst, gSrc);
  async_ld_b128(ldsDst + 8, gSrc + 8);
}
__device__ __forceinline__ void async_row32(_Float16* ldsDst,
                                            const _Float16* gSrc) {
#pragma unroll
  for (int i = 0; i < 4; ++i) async_ld_b128(ldsDst + i * 8, gSrc + i * 8);
}

// ---------------------------------------------------------------------------
// Lane-xor max reduction within 16-lane halves via ds_swizzle.
// One xor step applied to all 8 independent rows at once so 8 swizzles are
// in flight per wait point instead of 1.
// ---------------------------------------------------------------------------
template <int PAT>
__device__ __forceinline__ float swz(float v) {
  return __int_as_float(__builtin_amdgcn_ds_swizzle(__float_as_int(v), PAT));
}
template <int PAT>
__device__ __forceinline__ void swz_max_step8(float (&v)[8]) {
  float t[8];
#pragma unroll
  for (int e = 0; e < 8; ++e) t[e] = swz<PAT>(v[e]);
#pragma unroll
  for (int e = 0; e < 8; ++e) v[e] = fmaxf(v[e], t[e]);
}
__device__ __forceinline__ void max_half16_x8(float (&v)[8]) {
  swz_max_step8<0x041f>(v);   // xor 1
  swz_max_step8<0x081f>(v);   // xor 2
  swz_max_step8<0x101f>(v);   // xor 4
  swz_max_step8<0x201f>(v);   // xor 8
}

// ---------------------------------------------------------------------------
// WMMA helpers (wave32, v_wmma_f32_16x16x32_f16)
// ---------------------------------------------------------------------------
__device__ __forceinline__ v8f wmma16(v16h a, v16h b, v8f c) {
  return __builtin_amdgcn_wmma_f32_16x16x32_f16(
      false, a, false, b, (short)0, c, false, false);
}

// A fragment (16 rows x 32 K): lane<16 -> row=lane, K={k0..+7, k0+16..+23};
// lane>=16 -> row=lane-16, K={k0+8..+15, k0+24..+31}
__device__ __forceinline__ v16h load_a_frag(const _Float16* base, int ldh,
                                            int k0, int lane) {
  const int r  = lane & 15;
  const int hi = lane >> 4;
  const _Float16* p = base + r * ldh + k0 + hi * 8;
  v8h lo  = *(const v8h*)(p);
  v8h hi8 = *(const v8h*)(p + 16);
  v16h out;
#pragma unroll
  for (int i = 0; i < 8; ++i) { out[i] = lo[i]; out[i + 8] = hi8[i]; }
  return out;
}

// B fragment from B^T rows (row n = column n of B):
// lane<16 -> col=lane, K={k0..+7, k0+16..+23} ; lane>=16 -> col=lane-16, K={k0+8..+15, k0+24..+31}
__device__ __forceinline__ v16h load_b_frag(const _Float16* baseT, int ldh,
                                            int k0, int lane) {
  const int n  = lane & 15;
  const int hi = lane >> 4;
  const _Float16* p = baseT + n * ldh + k0 + hi * 8;
  v8h a = *(const v8h*)(p);
  v8h b = *(const v8h*)(p + 16);
  v16h out;
#pragma unroll
  for (int i = 0; i < 8; ++i) { out[i] = a[i]; out[i + 8] = b[i]; }
  return out;
}

// ---------------------------------------------------------------------------
// f32 -> f16 cast, 4 elements per thread
// ---------------------------------------------------------------------------
__global__ void cvt_f32_f16(const float* __restrict__ s,
                            _Float16* __restrict__ d, int n4) {
  int i = blockIdx.x * blockDim.x + threadIdx.x;
  if (i < n4) {
    float4 v = ((const float4*)s)[i];
    v4h o;
    o[0] = (_Float16)v.x; o[1] = (_Float16)v.y;
    o[2] = (_Float16)v.z; o[3] = (_Float16)v.w;
    ((v4h*)d)[i] = o;
  }
}

// ---------------------------------------------------------------------------
// Y[m][n] = scale * sum_k A[m][k]*W[n][k]  (+bias, f32 out if OUT_F32)
// 128x128 block tile, 8 waves in 4(M)x2(N) grid, wave tile 32x64,
// K-step 64, double-buffered LDS filled by async-to-LDS copies.
// ---------------------------------------------------------------------------
template <bool OUT_F32>
__global__ __launch_bounds__(256) void gemm_wt(
    const _Float16* __restrict__ A,   // [M][K]
    const _Float16* __restrict__ W,   // [N][K]
    void* __restrict__ Yv,            // [M][N] f16 or f32
    const float* __restrict__ bias,   // [N] (OUT_F32 only)
    int N, int K, float scale) {
  __shared__ __align__(16) _Float16 sA[2][128 * 64];
  __shared__ __align__(16) _Float16 sB[2][128 * 64];
  const int tid  = threadIdx.x;
  const int lane = tid & 31;
  const int wave = tid >> 5;
  const int wm = wave & 3;          // 0..3 : M sub-block
  const int wn = wave >> 2;         // 0..1 : N sub-block
  const int mBase = blockIdx.y * 128;
  const int nBase = blockIdx.x * 128;
  const int sr = tid >> 1;          // staging row 0..127
  const int sc = (tid & 1) * 32;    // staging col 0 or 32

  auto stage = [&](int buf, int k0) {
    async_row32(&sA[buf][sr * 64 + sc], &A[(mBase + sr) * K + k0 + sc]);
    async_row32(&sB[buf][sr * 64 + sc], &W[(nBase + sr) * K + k0 + sc]);
  };

  v8f acc[2][4];
#pragma unroll
  for (int mi = 0; mi < 2; ++mi)
#pragma unroll
    for (int ni = 0; ni < 4; ++ni) acc[mi][ni] = v8f{};

  const int KT = K / 64;
  stage(0, 0);
  for (int kt = 0; kt < KT; ++kt) {
    const int cur = kt & 1;
    wait_async0();          // our async writes for buffer `cur` complete
    __syncthreads();        // everyone's writes visible; prev reads of cur^1 done
    if (kt + 1 < KT) stage(cur ^ 1, (kt + 1) * 64);

    const _Float16* aBase = &sA[cur][(wm * 32) * 64];
    v16h af[2][2];
#pragma unroll
    for (int mi = 0; mi < 2; ++mi)
#pragma unroll
      for (int kc = 0; kc < 2; ++kc)
        af[mi][kc] = load_a_frag(aBase + mi * 16 * 64, 64, kc * 32, lane);
#pragma unroll
    for (int ni = 0; ni < 4; ++ni) {
      const _Float16* bBase = &sB[cur][(wn * 64 + ni * 16) * 64];
      v16h bf0 = load_b_frag(bBase, 64, 0, lane);
      v16h bf1 = load_b_frag(bBase, 64, 32, lane);
      acc[0][ni] = wmma16(af[0][0], bf0, acc[0][ni]);
      acc[0][ni] = wmma16(af[0][1], bf1, acc[0][ni]);
      acc[1][ni] = wmma16(af[1][0], bf0, acc[1][ni]);
      acc[1][ni] = wmma16(af[1][1], bf1, acc[1][ni]);
    }
  }

  const int half = lane >> 4, ln = lane & 15;
#pragma unroll
  for (int mi = 0; mi < 2; ++mi) {
    const int row0 = mBase + wm * 32 + mi * 16 + half * 8;
#pragma unroll
    for (int ni = 0; ni < 4; ++ni) {
      const int col = nBase + wn * 64 + ni * 16 + ln;
#pragma unroll
      for (int e = 0; e < 8; ++e) {
        if (OUT_F32)
          ((float*)Yv)[(row0 + e) * N + col] = acc[mi][ni][e] + bias[col];
        else
          ((_Float16*)Yv)[(row0 + e) * N + col] =
              (_Float16)(acc[mi][ni][e] * scale);
      }
    }
  }
}

// ---------------------------------------------------------------------------
// Flash attention with relative-position bias (Q pre-scaled by hd^-0.5).
// Block = (batch*head, 128-query tile), 8 waves, 16 query rows per wave.
// K staged async, V transposed via regs, bias band staged to LDS.
// Row max: batched ds_swizzle xor-reduce.  Row sum: WMMA against ones.
// ---------------------------------------------------------------------------
__global__ __launch_bounds__(256) void attn_kernel(
    const _Float16* __restrict__ Q,       // [B*S][D_MODEL]
    const _Float16* __restrict__ Km,      // [B*S][D_MODEL]
    const _Float16* __restrict__ V,       // [B*S][D_MODEL]
    const float* __restrict__ bias_table, // [2*SEQ-1][NUM_HEADS]
    _Float16* __restrict__ O) {           // [B*S][D_MODEL]
  const int bh = blockIdx.x;
  const int b = bh >> 4, h = bh & 15;
  const int qBase = blockIdx.y * 128;
  const int tid = threadIdx.x, lane = tid & 31, wave = tid >> 5;
  const int half = lane >> 4, ln = lane & 15;

  __shared__ __align__(16) _Float16 sQ[128 * 64];     // 16 KB
  __shared__ __align__(16) _Float16 sK[2][64 * 64];   // 16 KB
  __shared__ __align__(16) _Float16 sVT[2][64 * 64];  // 16 KB, v^T : [d][k]
  __shared__ __align__(16) _Float16 sP[8][16 * 64];   // 16 KB, per-wave P
  __shared__ float sBias[2][192];                     // 1.5 KB

  const int qr = tid >> 1;          // q staging row 0..127
  const int qc = (tid & 1) * 32;    // q staging col 0 or 32
  const int kr = tid >> 2;          // k/v staging row 0..63
  const int kc = (tid & 3) * 16;    // k/v staging col 0,16,32,48

  auto stage_kv = [&](int buf, int kt) {
    const _Float16* krow = Km + (b * SEQ + kt * 64) * D_MODEL + h * HD;
    const _Float16* vrow = V  + (b * SEQ + kt * 64) * D_MODEL + h * HD;
    async_row16(&sK[buf][kr * 64 + kc], &krow[kr * D_MODEL + kc]);
#pragma unroll
    for (int i = 0; i < 2; ++i) {
      v8h vv = *(const v8h*)&vrow[kr * D_MODEL + kc + i * 8];
#pragma unroll
      for (int u = 0; u < 8; ++u)
        sVT[buf][(kc + i * 8 + u) * 64 + kr] = vv[u];  // transpose on store
    }
    // bias diagonal band: rel = m-n+2047, m in [qBase,qBase+128),
    // n in [kt*64, kt*64+64)  ->  191 values starting at relMin
    if (tid < 192) {
      int rel = qBase - kt * 64 + (SEQ - 1) - 63 + tid;
      if (rel > 2 * SEQ - 2) rel = 2 * SEQ - 2;
      sBias[buf][tid] = bias_table[rel * NUM_HEADS + h];
    }
  };

  // prologue: stage Q (async) and first K/V tile
  const _Float16* qrow = Q + (b * SEQ + qBase) * D_MODEL + h * HD;
  async_row32(&sQ[qr * 64 + qc], &qrow[qr * D_MODEL + qc]);
  stage_kv(0, 0);
  wait_async0();
  __syncthreads();

  const v16h aq0 = load_a_frag(&sQ[wave * 16 * 64], 64, 0, lane);
  const v16h aq1 = load_a_frag(&sQ[wave * 16 * 64], 64, 32, lane);

  v16h ones;
#pragma unroll
  for (int i = 0; i < 16; ++i) ones[i] = (_Float16)1.0f;

  v8f acc[4] = {v8f{}, v8f{}, v8f{}, v8f{}};
  float rmax[8], rsum[8];
#pragma unroll
  for (int e = 0; e < 8; ++e) { rmax[e] = -1e30f; rsum[e] = 0.0f; }

  const int KT = SEQ / 64;
  for (int kt = 0; kt < KT; ++kt) {
    const int cur = kt & 1;
    if (kt + 1 < KT) stage_kv(cur ^ 1, kt + 1);   // overlap with compute

    // ---- scores S = q . k^T (16x64 per wave) ----
    v8f s[4];
#pragma unroll
    for (int j = 0; j < 4; ++j) {
      v16h bk0 = load_b_frag(&sK[cur][j * 16 * 64], 64, 0, lane);
      v16h bk1 = load_b_frag(&sK[cur][j * 16 * 64], 64, 32, lane);
      v8f z = v8f{};
      z = wmma16(aq0, bk0, z);
      s[j] = wmma16(aq1, bk1, z);
    }

    // ---- relative-position bias from LDS band ----
    float cmax[8];
#pragma unroll
    for (int e = 0; e < 8; ++e) cmax[e] = -1e30f;
#pragma unroll
    for (int j = 0; j < 4; ++j) {
      const int nloc = j * 16 + ln;
#pragma unroll
      for (int e = 0; e < 8; ++e) {
        const int mloc = wave * 16 + half * 8 + e;
        float sv = s[j][e] + sBias[cur][mloc - nloc + 63];
        s[j][e] = sv;
        cmax[e] = fmaxf(cmax[e], sv);
      }
    }

    // ---- batched row-max reduction (8 rows, 4 xor steps) ----
    max_half16_x8(cmax);

    // ---- online softmax: P -> LDS, rescale accumulators ----
    float cf[8];
#pragma unroll
    for (int e = 0; e < 8; ++e) {
      float nm = fmaxf(rmax[e], cmax[e]);
      cf[e] = __expf(rmax[e] - nm);
      rmax[e] = nm;
      const int mrow = half * 8 + e;
#pragma unroll
      for (int j = 0; j < 4; ++j)
        sP[wave][mrow * 64 + j * 16 + ln] = (_Float16)__expf(s[j][e] - nm);
#pragma unroll
      for (int d = 0; d < 4; ++d) acc[d][e] *= cf[e];
    }
    asm volatile("s_wait_dscnt 0x0" ::: "memory");  // wave-local LDS RAW fence

    // ---- P fragments (A layout) ----
    v16h pa0 = load_a_frag(sP[wave], 64, 0, lane);
    v16h pa1 = load_a_frag(sP[wave], 64, 32, lane);

    // ---- row sums via WMMA: rowsum(P) = P x ones, broadcast in C layout ----
    v8f rs = v8f{};
    rs = wmma16(pa0, ones, rs);
    rs = wmma16(pa1, ones, rs);
#pragma unroll
    for (int e = 0; e < 8; ++e) rsum[e] = rsum[e] * cf[e] + rs[e];

    // ---- ctx += P . V ----
#pragma unroll
    for (int d = 0; d < 4; ++d) {
      v16h bv0 = load_b_frag(&sVT[cur][d * 16 * 64], 64, 0, lane);
      v16h bv1 = load_b_frag(&sVT[cur][d * 16 * 64], 64, 32, lane);
      acc[d] = wmma16(pa0, bv0, acc[d]);
      acc[d] = wmma16(pa1, bv1, acc[d]);
    }

    wait_async0();      // next K tile resident (our lanes)
    __syncthreads();    // all waves done with `cur`, next buffers complete
  }

  // ---- normalize and store ctx (f16) ----
#pragma unroll
  for (int e = 0; e < 8; ++e) {
    float inv = 1.0f / rsum[e];
    int m = qBase + wave * 16 + half * 8 + e;
#pragma unroll
    for (int d = 0; d < 4; ++d)
      O[(b * SEQ + m) * D_MODEL + h * HD + d * 16 + ln] =
          (_Float16)(acc[d][e] * inv);
  }
}

// ---------------------------------------------------------------------------
// Host-side orchestration
// ---------------------------------------------------------------------------
extern "C" void kernel_launch(void* const* d_in, const int* in_sizes, int n_in,
                              void* d_out, int out_size, void* d_ws,
                              size_t ws_size, hipStream_t stream) {
  const float* x          = (const float*)d_in[0];
  const float* Wq         = (const float*)d_in[1];
  const float* Wk         = (const float*)d_in[2];
  const float* Wv         = (const float*)d_in[3];
  const float* Wo         = (const float*)d_in[4];
  const float* bo         = (const float*)d_in[5];
  const float* bias_table = (const float*)d_in[6];
  float* out = (float*)d_out;

  const size_t NX = (size_t)MTOT * D_MODEL;       // 8M elems
  const size_t NW = (size_t)D_MODEL * D_MODEL;    // 1M elems
  _Float16* xh  = (_Float16*)d_ws;
  _Float16* wqh = xh  + NX;
  _Float16* wkh = wqh + NW;
  _Float16* wvh = wkh + NW;
  _Float16* woh = wvh + NW;
  _Float16* qh  = woh + NW;
  _Float16* kh  = qh  + NX;
  _Float16* vh  = kh  + NX;
  _Float16* ch  = vh  + NX;

  {
    int n4 = (int)(NX / 4);
    cvt_f32_f16<<<(n4 + 255) / 256, 256, 0, stream>>>(x, xh, n4);
    n4 = (int)(NW / 4);
    cvt_f32_f16<<<(n4 + 255) / 256, 256, 0, stream>>>(Wq, wqh, n4);
    cvt_f32_f16<<<(n4 + 255) / 256, 256, 0, stream>>>(Wk, wkh, n4);
    cvt_f32_f16<<<(n4 + 255) / 256, 256, 0, stream>>>(Wv, wvh, n4);
    cvt_f32_f16<<<(n4 + 255) / 256, 256, 0, stream>>>(Wo, woh, n4);
  }

  dim3 gg(D_MODEL / 128, MTOT / 128);   // (8, 64)
  dim3 bb(256);
  const float scale = 0.125f;           // hd^-0.5, folded into Q
  gemm_wt<false><<<gg, bb, 0, stream>>>(xh, wqh, qh, nullptr,
                                        D_MODEL, D_MODEL, scale);
  gemm_wt<false><<<gg, bb, 0, stream>>>(xh, wkh, kh, nullptr,
                                        D_MODEL, D_MODEL, 1.0f);
  gemm_wt<false><<<gg, bb, 0, stream>>>(xh, wvh, vh, nullptr,
                                        D_MODEL, D_MODEL, 1.0f);

  attn_kernel<<<dim3(BATCH * NUM_HEADS, SEQ / 128), bb, 0, stream>>>(
      qh, kh, vh, bias_table, ch);

  gemm_wt<true><<<gg, bb, 0, stream>>>(ch, woh, out, bo,
                                       D_MODEL, D_MODEL, 1.0f);
}
